// rnn_two_20048907337989
// MI455X (gfx1250) — compile-verified, matrix operand order checked by reading, not run on hardware
//
#include <hip/hip_runtime.h>

#define BATCH   8192
#define TSTEPS  16
#define NN      256
#define BM      64          // batch rows per workgroup
#define NTHREADS 256        // 8 wave32
#define HSTRIDE 264         // padded H row stride (halves): 528B -> bank-conflict-free

typedef __attribute__((ext_vector_type(16))) _Float16 v16h;
typedef __attribute__((ext_vector_type(8)))  _Float16 v8h;
typedef __attribute__((ext_vector_type(8)))  float    v8f;
typedef __attribute__((ext_vector_type(4)))  float    v4f;

// exact pointee type expected by the async-to-LDS builtin (per hipcc diag)
typedef __attribute__((__vector_size__(4 * sizeof(int)))) int vsi4;
typedef __attribute__((address_space(1))) vsi4 g_vsi4;   // global
typedef __attribute__((address_space(3))) vsi4 l_vsi4;   // LDS

#define HAS_ASYNC_LDS __has_builtin(__builtin_amdgcn_global_load_async_to_lds_b128)

// ---------------------------------------------------------------------------
// Weight repack: 5 NxN f32 matrices -> f16 in WMMA B-fragment order.
// Per matrix layout: [kc(8)][nt(16)][p(2)][lane(32)][ii(8)] halves (64K halves).
// Plane p holds frag halves 8p..8p+7, so a B-frag ds_load_b128 has lanes at a
// contiguous 16B stride (no bank conflicts beyond the 2-cycle minimum).
// ---------------------------------------------------------------------------
__global__ void repack_weights(const float* __restrict__ W1,
                               const float* __restrict__ U1,
                               const float* __restrict__ W2,
                               const float* __restrict__ U2,
                               const float* __restrict__ Wo,
                               _Float16* __restrict__ dst) {
  int o = blockIdx.x * blockDim.x + threadIdx.x;
  if (o >= 5 * 65536) return;
  int m    = o >> 16;
  int r    = o & 65535;
  int ii   = r & 7;
  int lane = (r >> 3) & 31;
  int p    = (r >> 8) & 1;
  int nt   = (r >> 9) & 15;
  int kc   = (r >> 13) & 7;
  int k = kc * 32 + ((lane >> 4) & 1) * 8 + ii + p * 16;
  int n = nt * 16 + (lane & 15);
  const float* src = (m == 0) ? W1 : (m == 1) ? U1 : (m == 2) ? W2
                   : (m == 3) ? U2 : Wo;
  dst[o] = (_Float16)src[k * NN + n];
}

// ---- fragment helpers ------------------------------------------------------

__device__ __forceinline__ v16h ldsA(const _Float16* H, int mrow, int kc, int half) {
  const _Float16* p = H + mrow * HSTRIDE + kc * 32 + half * 8;
  v8h lo = *(const v8h*)p;
  v8h hi = *(const v8h*)(p + 16);
  v16h a;
#pragma unroll
  for (int i = 0; i < 8; i++) { a[i] = lo[i]; a[i + 8] = hi[i]; }
  return a;
}

__device__ __forceinline__ v16h gxA(const float* xr, int kc, int half) {
  const float* p = xr + kc * 32 + half * 8;
  v4f f0 = *(const v4f*)p;
  v4f f1 = *(const v4f*)(p + 4);
  v4f f2 = *(const v4f*)(p + 16);
  v4f f3 = *(const v4f*)(p + 20);
  v16h a;
#pragma unroll
  for (int i = 0; i < 4; i++) {
    a[i]      = (_Float16)f0[i];
    a[i + 4]  = (_Float16)f1[i];
    a[i + 8]  = (_Float16)f2[i];
    a[i + 12] = (_Float16)f3[i];
  }
  return a;
}

// B fragment: two 16B planes at +0 and +512B.
__device__ __forceinline__ v16h ldB(const _Float16* pb, int j) {
  v8h lo = *(const v8h*)(pb + j * 512);
  v8h hi = *(const v8h*)(pb + j * 512 + 256);
  v16h b;
#pragma unroll
  for (int i = 0; i < 8; i++) { b[i] = lo[i]; b[i + 8] = hi[i]; }
  return b;
}

// Stage one 16KB fragment-ordered weight chunk into LDS (async DMA, 4 ops/wave).
__device__ __forceinline__ void stage_chunk(const _Float16* __restrict__ src,
                                            _Float16* __restrict__ dst, int tid) {
#if HAS_ASYNC_LDS
  const char* s = (const char*)src + tid * 16;
  char* d = (char*)dst + tid * 16;
#pragma unroll
  for (int i = 0; i < 4; i++) {
    __builtin_amdgcn_global_load_async_to_lds_b128(
        (g_vsi4*)(s + i * 4096), (l_vsi4*)(d + i * 4096), 0, 0);
  }
#else
  const v8h* s = (const v8h*)src;
  v8h* d = (v8h*)dst;
#pragma unroll
  for (int i = 0; i < 4; i++) d[tid + i * NTHREADS] = s[tid + i * NTHREADS];
#endif
}

template <int N>
__device__ __forceinline__ void wait_async_lds() {
#if HAS_ASYNC_LDS
#if __has_builtin(__builtin_amdgcn_s_wait_asynccnt)
  __builtin_amdgcn_s_wait_asynccnt(N);
#else
  if (N == 0) asm volatile("s_wait_asynccnt 0x0" ::: "memory");
  else        asm volatile("s_wait_asynccnt 0x4" ::: "memory");
#endif
#endif
}

// 8 WMMAs against one staged chunk; B fragments software-pipelined 4 deep.
__device__ __forceinline__ void mma_tiles(v8f acc[8], v16h a,
                                          const _Float16* cur, int lane, int nh) {
  const _Float16* pb = cur + nh * 4096 + lane * 8;
  v16h b[4];
#pragma unroll
  for (int jj = 0; jj < 4; jj++) b[jj] = ldB(pb, jj);
#pragma unroll
  for (int j = 0; j < 8; j++) {
    v16h bb = b[j & 3];
    if (j < 4) b[j & 3] = ldB(pb, j + 4);
    acc[j] = __builtin_amdgcn_wmma_f32_16x16x32_f16(
        false, a, false, bb, (short)0, acc[j], false, false);
  }
}

// Pipelined multi-chunk GEMM accumulate: triple-buffered async weight staging
// (prefetch distance 2) and A-fragments software-pipelined one chunk ahead so
// the post-barrier critical path is only the B-fragment loads.
// A source: chunk<8 -> x (if X_FIRST) else HA; chunk>=8 -> HB.
template <int NCHUNK, bool X_FIRST>
__device__ __forceinline__ void gemm2(v8f acc[8],
                                      const _Float16* __restrict__ wbase,
                                      const _Float16* __restrict__ HA,
                                      const _Float16* __restrict__ HB,
                                      const float* __restrict__ xr,
                                      _Float16* __restrict__ sB,  // 3 x 8192 halves
                                      int tid, int lane, int mrow, int half,
                                      int nh) {
  int cur = 0, nxt = 8192, nx2 = 16384;
  // Prologue writes are fenced by the caller's preceding __syncthreads().
  stage_chunk(wbase, sB + cur, tid);
  stage_chunk(wbase + 8192, sB + nxt, tid);
  v16h a = X_FIRST ? gxA(xr, 0, half) : ldsA(HA, mrow, 0, half);

  constexpr int P1END = (NCHUNK == 8) ? 7 : 8;
#pragma unroll 1
  for (int cc = 0; cc < P1END; cc++) {
    wait_async_lds<4>();                // chunk cc landed (next stays in flight)
    __syncthreads();
    if (NCHUNK > 8 || cc < 6)
      stage_chunk(wbase + (cc + 2) * 8192, sB + nx2, tid);
    v16h acur = a;                      // prefetch A for chunk cc+1 (H/x stable)
    a = (X_FIRST && cc + 1 < 8) ? gxA(xr, cc + 1, half)
      : ldsA((cc + 1 < 8) ? HA : HB, mrow, (cc + 1) & 7, half);
    mma_tiles(acc, acur, sB + cur, lane, nh);
    int tmp = cur; cur = nxt; nxt = nx2; nx2 = tmp;
  }
#pragma unroll 1
  for (int cc = 8; cc < NCHUNK - 1; cc++) {   // NCHUNK=16: chunks 8..14
    wait_async_lds<4>();
    __syncthreads();
    if (cc < 14)
      stage_chunk(wbase + (cc + 2) * 8192, sB + nx2, tid);
    v16h acur = a;
    a = ldsA(HB, mrow, (cc + 1) & 7, half);
    mma_tiles(acc, acur, sB + cur, lane, nh);
    int tmp = cur; cur = nxt; nxt = nx2; nx2 = tmp;
  }
  // final chunk: drain everything
  wait_async_lds<0>();
  __syncthreads();
  mma_tiles(acc, a, sB + cur, lane, nh);
}

// Scatter f32 accumulators into a padded row-major f16 H tile in LDS.
__device__ __forceinline__ void store_h(const v8f acc[8], _Float16* H,
                                        int mt, int nh, int lane) {
  int mbase = mt * 16 + ((lane >> 4) & 1) * 8;
  int ncol  = lane & 15;
#pragma unroll
  for (int j = 0; j < 8; j++) {
    int n = (nh * 8 + j) * 16 + ncol;
#pragma unroll
    for (int v = 0; v < 8; v++) H[(mbase + v) * HSTRIDE + n] = (_Float16)acc[j][v];
  }
}

// ---------------------------------------------------------------------------
// Fused 2-layer linear RNN over all T steps for a 64-row batch tile.
// ---------------------------------------------------------------------------
__global__ __launch_bounds__(NTHREADS) void rnn_fused(
    const float* __restrict__ x, const _Float16* __restrict__ wf,
    float* __restrict__ out) {
  __shared__ __align__(32) _Float16 sH1[BM * HSTRIDE];  // 33 KB, padded rows
  __shared__ __align__(32) _Float16 sH2[BM * HSTRIDE];  // 33 KB
  __shared__ __align__(32) _Float16 sB[3 * 8192];       // 3 x 16 KB staging

  const int tid  = threadIdx.x;
  const int lane = tid & 31;
  const int wave = tid >> 5;
  const int mt   = wave & 3;               // M tile (16 rows) in WG tile
  const int nh   = wave >> 2;              // which 128-column half of N
  const int half = (lane >> 4) & 1;
  const int mrow = mt * 16 + (lane & 15);  // this lane's A-fragment row
  const int b0   = blockIdx.x * BM;

  const _Float16* L1f = wf;                // W1(8 chunks) then U1(8 chunks)
  const _Float16* L2f = wf + 2 * 65536;    // W2 then U2
  const _Float16* Wof = wf + 4 * 65536;    // Wo

  {
    v8h z = {};
    for (int idx = tid; idx < BM * HSTRIDE / 8; idx += NTHREADS) {
      ((v8h*)sH1)[idx] = z;
      ((v8h*)sH2)[idx] = z;
    }
  }
  __syncthreads();

  const float* xrow = x + (size_t)(b0 + mrow) * (TSTEPS * NN);

#pragma unroll 1
  for (int t = 0; t < TSTEPS; t++) {
    const float* xr = xrow + t * NN;

    // h1_new = x_t @ W1 + h1 @ U1
    v8f acc[8];
#pragma unroll
    for (int j = 0; j < 8; j++) acc[j] = (v8f){};
    gemm2<16, true>(acc, L1f, sH1, sH1, xr, sB, tid, lane, mrow, half, nh);
    __syncthreads();                  // all reads of old h1 done
    store_h(acc, sH1, mt, nh, lane);
    __syncthreads();                  // new h1 visible to all waves

    // h2_new = h1_new @ W2 + h2 @ U2
    v8f acc2[8];
#pragma unroll
    for (int j = 0; j < 8; j++) acc2[j] = (v8f){};
    gemm2<16, false>(acc2, L2f, sH1, sH2, xr, sB, tid, lane, mrow, half, nh);
    __syncthreads();
    store_h(acc2, sH2, mt, nh, lane);
    __syncthreads();
  }

  // out = h2_last @ Wo   (f32 straight to global)
  v8f acc[8];
#pragma unroll
  for (int j = 0; j < 8; j++) acc[j] = (v8f){};
  gemm2<8, false>(acc, Wof, sH2, sH2, nullptr, sB, tid, lane, mrow, half, nh);

  const int mbase = mt * 16 + half * 8;
  const int ncol  = lane & 15;
#pragma unroll
  for (int j = 0; j < 8; j++) {
    int n = (nh * 8 + j) * 16 + ncol;
#pragma unroll
    for (int v = 0; v < 8; v++)
      out[(size_t)(b0 + mbase + v) * NN + n] = acc[j][v];
  }
}

extern "C" void kernel_launch(void* const* d_in, const int* in_sizes, int n_in,
                              void* d_out, int out_size, void* d_ws, size_t ws_size,
                              hipStream_t stream) {
  const float* x  = (const float*)d_in[0];
  const float* W1 = (const float*)d_in[1];
  const float* U1 = (const float*)d_in[2];
  const float* W2 = (const float*)d_in[3];
  const float* U2 = (const float*)d_in[4];
  const float* Wo = (const float*)d_in[5];
  _Float16* wf = (_Float16*)d_ws;          // needs 5*65536*2 = 640 KB scratch
  float* out = (float*)d_out;

  repack_weights<<<(5 * 65536 + 255) / 256, 256, 0, stream>>>(W1, U1, W2, U2, Wo, wf);
  rnn_fused<<<BATCH / BM, NTHREADS, 0, stream>>>(x, wf, out);
}
